// TempCtxAttentionNS_70042326663621
// MI455X (gfx1250) — matrix operand order repaired and checked by reading, not compiled, not run on hardware
//
#include <hip/hip_runtime.h>
#include <hip/hip_bf16.h>
#include <math.h>

// ---------------------------------------------------------------------------
// CDNA5 (gfx1250) implementation:
//  * all GEMMs via v_wmma_f32_16x16x32_bf16 (bf16 inputs, fp32 accum)
//    - LDS layouts chosen so WMMA fragments load as ds_load_b128
//    - 2 WMMAs per wave per K-step (A-fragment reuse across two N tiles)
//    - attention intermediate X kept in bf16; bf16-A staging uses
//      GLOBAL_LOAD_ASYNC_TO_LDS_B64 (ASYNCcnt) when the toolchain exposes it
//  * two-stage feature-axis softmax as wave32 shuffle-reduction kernel
// ---------------------------------------------------------------------------

typedef __attribute__((ext_vector_type(2)))  int    v2i;
typedef __attribute__((ext_vector_type(4)))  __bf16 v4bf;
typedef __attribute__((ext_vector_type(8)))  __bf16 v8bf;
typedef __attribute__((ext_vector_type(16))) __bf16 v16bf;
typedef __attribute__((ext_vector_type(4)))  float  v4f;
typedef __attribute__((ext_vector_type(8)))  float  v8f;

#if defined(__has_builtin)
#  if __has_builtin(__builtin_amdgcn_global_load_async_to_lds_b64)
#    define HAVE_ASYNC_LDS 1
#  else
#    define HAVE_ASYNC_LDS 0
#  endif
#else
#  define HAVE_ASYNC_LDS 0
#endif

#define TILE_M 32
#define TILE_N 128
#define KSTEP  32

__device__ __forceinline__ void wait_async_all() {
#if HAVE_ASYNC_LDS
#  if __has_builtin(__builtin_amdgcn_s_wait_asynccnt)
    __builtin_amdgcn_s_wait_asynccnt(0);
#  else
    asm volatile("s_wait_asynccnt 0" ::: "memory");
#  endif
#endif
}

// C[M,N] = A[M,K] @ B[K,N] + bias. A is fp32 or bf16 (template), B fp32,
// C fp32. 256 threads = 8 waves; each wave computes a 16x32 output slab
// (2 WMMAs, shared A fragment). Requires N % 128 == 0, K % 32 == 0.
// For TA == __bf16, additionally requires M % 32 == 0 (no row masking on the
// async DMA path).
template <typename TA>
__global__ __launch_bounds__(256) void gemm_bf16_wmma(
    const TA* __restrict__ A, const float* __restrict__ B,
    const float* __restrict__ bias, float* __restrict__ C,
    int M, int N, int K)
{
    // sA row-major: fragment reads are 16B-contiguous runs per lane.
    // sB stored TRANSPOSED [n][k]: fragment reads are 32B-contiguous per lane.
    __shared__ __bf16 sA[TILE_M][KSTEP];   // 2 KB
    __shared__ __bf16 sBt[TILE_N][KSTEP];  // 8 KB

    const int tid  = threadIdx.x;
    const int lane = tid & 31;
    const int wave = tid >> 5;        // 0..7
    const int wm   = wave >> 2;       // 0..1 : 16-row group
    const int wn   = wave & 3;        // 0..3 : 32-col group

    const int blockRow = blockIdx.y * TILE_M;
    const int blockCol = blockIdx.x * TILE_N;

    const int hl = lane >> 4;         // lane half: 0 = lanes 0-15, 1 = 16-31
    const int rm = lane & 15;

    // staging assignments (branchless, fixed work per thread)
    const int ar = tid >> 3;              // A: row 0..31
    const int ac = (tid & 7) * 4;         // A: 4 consecutive K values
    const int agr = blockRow + ar;
    const int acr = (agr < M) ? agr : (M - 1);   // clamped row (valid address)
    const bool arow_ok = (agr < M);

    const int bn = tid & 127;             // B: column within tile, 0..127
    const int bk = (tid >> 7) * 16;       // B: K group base, 0 or 16

    v8f acc0 = {};
    v8f acc1 = {};

    for (int k0 = 0; k0 < K; k0 += KSTEP) {
        // ---- stage A tile (32x32 x bf16) ----
        if constexpr (sizeof(TA) == 4) {
            // fp32 source: 1 global b128 + packed cvt + 1 ds b64 per thread
            v4f a4 = *(const v4f*)(A + (size_t)acr * K + k0 + ac);
            v4bf p;
            p[0] = arow_ok ? (__bf16)a4[0] : (__bf16)0.0f;
            p[1] = arow_ok ? (__bf16)a4[1] : (__bf16)0.0f;
            p[2] = arow_ok ? (__bf16)a4[2] : (__bf16)0.0f;
            p[3] = arow_ok ? (__bf16)a4[3] : (__bf16)0.0f;
            *(v4bf*)&sA[ar][ac] = p;
        } else {
            // bf16 source: straight 8-byte copy global -> LDS
#if HAVE_ASYNC_LDS
            __builtin_amdgcn_global_load_async_to_lds_b64(
                (__attribute__((address_space(1))) v2i*)(A + (size_t)acr * K + k0 + ac),
                (__attribute__((address_space(3))) v2i*)&sA[ar][ac],
                0, 0);
#else
            *(v4bf*)&sA[ar][ac] = *(const v4bf*)(A + (size_t)acr * K + k0 + ac);
#endif
        }
        // ---- stage B tile (32x128) transposed: 16 coalesced b32 loads,
        //      packed bf16, 2 ds b128 stores / thread
        {
            const float* bsrc = B + (size_t)(k0 + bk) * N + blockCol + bn;
            v8bf p0, p1;
            #pragma unroll
            for (int j = 0; j < 8; ++j) p0[j] = (__bf16)bsrc[(size_t)j * N];
            #pragma unroll
            for (int j = 0; j < 8; ++j) p1[j] = (__bf16)bsrc[(size_t)(j + 8) * N];
            *(v8bf*)&sBt[bn][bk]     = p0;
            *(v8bf*)&sBt[bn][bk + 8] = p1;
        }
        // speculative prefetch of next K-slab (global_prefetch_b8)
        if (k0 + KSTEP < K) {
            __builtin_prefetch(A + (size_t)acr * K + k0 + KSTEP + ac, 0, 1);
            __builtin_prefetch(B + (size_t)(k0 + KSTEP + bk) * N + blockCol + bn, 0, 1);
        }
        if constexpr (sizeof(TA) == 2) wait_async_all();
        __syncthreads();

        // ---- fragments (k = hl*16 + i order, same for A and B), contiguous b128 LDS loads ----
        // A frag element i <-> K = hl*16 + i
        v8bf a_lo = *(const v8bf*)&sA[wm * 16 + rm][hl * 16];
        v8bf a_hi = *(const v8bf*)&sA[wm * 16 + rm][hl * 16 + 8];
        v16bf afrag = __builtin_shufflevector(a_lo, a_hi,
            0, 1, 2, 3, 4, 5, 6, 7, 8, 9, 10, 11, 12, 13, 14, 15);
        // B frag element i <-> K = hl*16 + i  (32B contiguous in sBt)
        v16bf bfrag0 = *(const v16bf*)&sBt[wn * 32 + rm][hl * 16];
        v16bf bfrag1 = *(const v16bf*)&sBt[wn * 32 + 16 + rm][hl * 16];

        acc0 = __builtin_amdgcn_wmma_f32_16x16x32_bf16(
            false, afrag, false, bfrag0, (short)0, acc0, false, false);
        acc1 = __builtin_amdgcn_wmma_f32_16x16x32_bf16(
            false, afrag, false, bfrag1, (short)0, acc1, false, false);
        __syncthreads();
    }

    // D layout: VGPR r -> row (r + 8*hl); lanes 0-15 / 16-31 both cover N 0-15
    const int col0 = blockCol + wn * 32 + rm;
    const int col1 = col0 + 16;
    const float b0 = bias[col0];
    const float b1 = bias[col1];
    #pragma unroll
    for (int r = 0; r < 8; ++r) {
        int row = blockRow + wm * 16 + r + hl * 8;
        if (row < M) {
            C[(size_t)row * N + col0] = acc0[r] + b0;
            C[(size_t)row * N + col1] = acc1[r] + b1;
        }
    }
}

// ---------------------------------------------------------------------------
// Fused two-stage feature-axis attention.
// One wave32 per (n,m,h). 64-element feature vectors live as 2 floats/lane.
// Stage 1: cv[d] = sum_q softmax_q(q[d]*k[q]/8) * v[q]
// Stage 2: out[d] = sum_q softmax_q(t[d]*cv[q]/8) * cv[q]
// Output X written in bf16 (sole consumer is the bf16 WMMA output GEMM).
// ---------------------------------------------------------------------------
__device__ __forceinline__ float wred_sum(float x) {
    #pragma unroll
    for (int o = 16; o > 0; o >>= 1) x += __shfl_xor(x, o, 32);
    return x;
}
__device__ __forceinline__ float wred_max(float x) {
    #pragma unroll
    for (int o = 16; o > 0; o >>= 1) x = fmaxf(x, __shfl_xor(x, o, 32));
    return x;
}

__global__ __launch_bounds__(256) void temp_ctx_attn(
    const float* __restrict__ qp,  // [48,512]  (n, h*64+d)
    const float* __restrict__ tp,  // [48,512]
    const float* __restrict__ kp,  // [32,512]  (m, h*64+q)
    const float* __restrict__ vp,  // [32,512]
    __bf16* __restrict__ X)        // [48*32, 512] bf16
{
    const int lane = threadIdx.x & 31;
    const int wave = threadIdx.x >> 5;
    const int idx  = blockIdx.x * 8 + wave;   // ((n*32)+m)*8 + h, < 12288
    const int h  = idx & 7;
    const int mm = (idx >> 3) & 31;
    const int nn = idx >> 8;

    const float scale = 0.125f;               // 1/sqrt(64)

    const float* qrow = qp + nn * 512 + h * 64;
    const float* trow = tp + nn * 512 + h * 64;
    const float* krow = kp + mm * 512 + h * 64;
    const float* vrow = vp + mm * 512 + h * 64;

    const float q0 = qrow[lane], q1 = qrow[lane + 32];
    const float t0 = trow[lane], t1 = trow[lane + 32];
    const float k0 = krow[lane], k1 = krow[lane + 32];
    const float v0 = vrow[lane], v1 = vrow[lane + 32];

    // ---------------- stage 1: context attention ----------------
    float cv0 = 0.0f, cv1 = 0.0f;
    for (int d = 0; d < 64; ++d) {
        float qd = (d < 32) ? __shfl(q0, d, 32) : __shfl(q1, d - 32, 32);
        float a  = qd * scale;
        float s0 = a * k0, s1 = a * k1;
        float mx = wred_max(fmaxf(s0, s1));
        float e0 = __expf(s0 - mx), e1 = __expf(s1 - mx);
        float den = wred_sum(e0 + e1);
        float num = wred_sum(e0 * v0 + e1 * v1);
        float cvd = num / den;
        if (lane == (d & 31)) { if (d < 32) cv0 = cvd; else cv1 = cvd; }
    }

    // ---------------- stage 2: temporal attention ----------------
    float o0 = 0.0f, o1 = 0.0f;
    for (int d = 0; d < 64; ++d) {
        float td = (d < 32) ? __shfl(t0, d, 32) : __shfl(t1, d - 32, 32);
        float a  = td * scale;
        float s0 = a * cv0, s1 = a * cv1;
        float mx = wred_max(fmaxf(s0, s1));
        float e0 = __expf(s0 - mx), e1 = __expf(s1 - mx);
        float den = wred_sum(e0 + e1);
        float num = wred_sum(e0 * cv0 + e1 * cv1);
        float od  = num / den;
        if (lane == (d & 31)) { if (d < 32) o0 = od; else o1 = od; }
    }

    __bf16* xrow = X + (size_t)(nn * 32 + mm) * 512 + h * 64;
    xrow[lane]      = (__bf16)o0;
    xrow[lane + 32] = (__bf16)o1;
}

// ---------------------------------------------------------------------------
extern "C" void kernel_launch(void* const* d_in, const int* in_sizes, int n_in,
                              void* d_out, int out_size, void* d_ws, size_t ws_size,
                              hipStream_t stream) {
    const float* query = (const float*)d_in[0];   // [48,256]
    const float* key   = (const float*)d_in[1];   // [32,512]
    const float* value = (const float*)d_in[2];   // [32,512]
    const float* timeM = (const float*)d_in[3];   // [48,64]
    const float* Wq = (const float*)d_in[4];  const float* bq = (const float*)d_in[5];
    const float* Wt = (const float*)d_in[6];  const float* bt = (const float*)d_in[7];
    const float* Wk = (const float*)d_in[8];  const float* bk = (const float*)d_in[9];
    const float* Wv = (const float*)d_in[10]; const float* bv = (const float*)d_in[11];
    const float* Wo = (const float*)d_in[12]; const float* bo = (const float*)d_in[13];
    float* out = (float*)d_out;                   // [48*32, 512]

    // workspace layout
    float*  ws  = (float*)d_ws;
    float*  qp  = ws;                 // 48*512 fp32
    float*  tpb = qp + 48 * 512;      // 48*512 fp32
    float*  kpb = tpb + 48 * 512;     // 32*512 fp32
    float*  vpb = kpb + 32 * 512;     // 32*512 fp32
    __bf16* Xbf = (__bf16*)(vpb + 32 * 512);  // 1536*512 bf16

    dim3 blk(256);

    // projections (N=512 -> grid.x = 512/128 = 4)
    gemm_bf16_wmma<float><<<dim3(4, 2), blk, 0, stream>>>(query, Wq, bq, qp, 48, 512, 256);
    gemm_bf16_wmma<float><<<dim3(4, 2), blk, 0, stream>>>(timeM, Wt, bt, tpb, 48, 512, 64);
    gemm_bf16_wmma<float><<<dim3(4, 1), blk, 0, stream>>>(key,   Wk, bk, kpb, 32, 512, 512);
    gemm_bf16_wmma<float><<<dim3(4, 1), blk, 0, stream>>>(value, Wv, bv, vpb, 32, 512, 512);

    // fused two-stage softmax: 12288 (n,m,h) waves, 8 waves / block
    temp_ctx_attn<<<dim3(1536), blk, 0, stream>>>(qp, tpb, kpb, vpb, Xbf);

    // output projection: [1536,512](bf16) @ [512,512] + bo   (M % 32 == 0)
    gemm_bf16_wmma<__bf16><<<dim3(4, 48), blk, 0, stream>>>(Xbf, Wo, bo, out, 1536, 512, 512);
}
